// CudaFastWeightDPFPTransformerLayer_15891378995479
// MI455X (gfx1250) — matrix operand order repaired and checked by reading, not compiled, not run on hardware
//
#include <hip/hip_runtime.h>
#include <hip/hip_bf16.h>
#include <math.h>

// ---------------------------------------------------------------------------
// CDNA5 (gfx1250) fast-weight DPFP transformer layer.
// wave32; WMMA bf16 16x16x32 for the two big GEMMs (16x32 C-tile per wave,
// dual independent accumulator chains for XDL latency hiding); LDS-resident
// delta-rule recurrence (64KB fast-weight state per (b,h) in LDS all 2048 steps).
// ---------------------------------------------------------------------------

typedef __bf16 bf16_t;
typedef __attribute__((ext_vector_type(16))) __bf16 v16bf;
typedef __attribute__((ext_vector_type(8)))  __bf16 v8bf;
typedef __attribute__((ext_vector_type(8)))  float  v8f;

#define SLEN   2048
#define BSZ    8
#define NHEAD  8
#define DHEAD  64
#define DMODEL 512
#define DK     256            // 2 * N_ROLL * DHEAD
#define EQ     1544           // NHEAD * (3*DHEAD + 1)
#define EQP    1568           // padded to multiple of 32 (49 * 32)
#define ROWS   (SLEN * BSZ)   // 16384
#define EPS_F  1e-5f
#define SCALE_F 0.125f        // 64^-0.5

// ---------------------------------------------------------------------------
// fp32 -> bf16 conversion with optional row padding (pad rows written as 0)
// ---------------------------------------------------------------------------
__global__ void cvt_pad_bf16(const float* __restrict__ in, bf16_t* __restrict__ out,
                             int rows_in, int rows_out, int cols) {
  size_t i = (size_t)blockIdx.x * blockDim.x + threadIdx.x;
  size_t tot = (size_t)rows_out * cols;
  if (i >= tot) return;
  int r = (int)(i / (size_t)cols);
  out[i] = (r < rows_in) ? (bf16_t)in[i] : (bf16_t)0.0f;
}

// ---------------------------------------------------------------------------
// C[M, Nreal] (f32) = A[M,K](bf16, row-major) * B[Npad,K](bf16, row-major)^T
// One 16x32 C tile per wave (two adjacent 16x16 N-tiles share the A fragment,
// independent accumulator chains); 8 waves = 256-wide N per block per M-tile.
// K-loop of v_wmma_f32_16x16x32_bf16, fully unrolled (32 WMMA per wave pass).
// Fragment layout used here (contiguous k, same order in A and B):
//   A 16x32 bf16 : lane<16 -> M=lane,    K = k..k+15
//                  lane>=16 -> M=lane-16, K = k+16..k+31
//   B 32x16 bf16 : lane<16 -> N=lane,    K = k..k+15
//                  lane>=16 -> N=lane-16, K = k+16..k+31
//   C 16x16 f32  : vgpr r -> M = r + 8*(lane>=16), N = lane%16
// ---------------------------------------------------------------------------
__global__ __launch_bounds__(256)
void gemm_bf16_wmma(const bf16_t* __restrict__ A, const bf16_t* __restrict__ B,
                    float* __restrict__ C, int M, int Nreal, int Npad, int K) {
  const int wave = threadIdx.x >> 5;
  const int lane = threadIdx.x & 31;
  const int pairTiles = Npad >> 5;       // 32-wide N chunks
  const int pair  = blockIdx.x * 8 + wave;
  const int tileM = blockIdx.y;
  if (pair >= pairTiles) return;         // uniform per wave: EXEC stays all-ones

  const int m0 = tileM << 4;
  const int n0 = pair << 5;
  const int half = lane >> 4;            // 0 | 1
  const int lr   = lane & 15;

  const bf16_t* arow  = A + (size_t)(m0 + lr) * K;
  const bf16_t* brow0 = B + (size_t)(n0 + lr) * K;
  const bf16_t* brow1 = B + (size_t)(n0 + 16 + lr) * K;

  v8f acc0 = {}, acc1 = {};
#pragma unroll
  for (int k = 0; k < DMODEL; k += 32) {
    v8bf alo = *reinterpret_cast<const v8bf*>(arow + k + half * 16);
    v8bf ahi = *reinterpret_cast<const v8bf*>(arow + k + 8 + half * 16);
    v16bf a;
#pragma unroll
    for (int i = 0; i < 8; ++i) { a[i] = alo[i]; a[8 + i] = ahi[i]; }
    v16bf b0 = *reinterpret_cast<const v16bf*>(brow0 + k + half * 16);
    v16bf b1 = *reinterpret_cast<const v16bf*>(brow1 + k + half * 16);
    acc0 = __builtin_amdgcn_wmma_f32_16x16x32_bf16(
        false, a, false, b0, (short)0, acc0, false, false);
    acc1 = __builtin_amdgcn_wmma_f32_16x16x32_bf16(
        false, a, false, b1, (short)0, acc1, false, false);
  }

  const int na = n0 + lr;
  const int nb = n0 + 16 + lr;
  if (na < Nreal) {
#pragma unroll
    for (int r = 0; r < 8; ++r) {
      int m = m0 + r + half * 8;
      C[(size_t)m * Nreal + na] = acc0[r];
    }
  }
  if (nb < Nreal) {
#pragma unroll
    for (int r = 0; r < 8; ++r) {
      int m = m0 + r + half * 8;
      C[(size_t)m * Nreal + nb] = acc1[r];
    }
  }
}

// ---------------------------------------------------------------------------
// DPFP featurize: per (l,b,h) item compute
//   y = [relu(x), relu(-x)]  (128)
//   z[(i-1)*128 + t] = y[t] * y[(t-i) mod 128], i in {1,2}  (256)
//   feat = z / sum(z)
// for both q and k; also sigmoid(beta) and copy v.
// One 256-thread block per item.
// ---------------------------------------------------------------------------
__global__ __launch_bounds__(256)
void dpfp_kernel(const float* __restrict__ qkvb,
                 float* __restrict__ qf, float* __restrict__ kf,
                 float* __restrict__ vv, float* __restrict__ beta) {
  const int item = blockIdx.x;        // l*64 + b*8 + h
  const int l = item >> 6;
  const int bh = item & 63;
  const int b = bh >> 3;
  const int h = bh & 7;
  const float* src = qkvb + (size_t)(l * BSZ + b) * EQ + h * (3 * DHEAD + 1);

  __shared__ float yq[128], yk[128], zq[256], zk[256], red[256], sums[2];
  const int t = threadIdx.x;

  if (t < 64) {
    float x = src[t];
    yq[t] = fmaxf(x, 0.0f); yq[t + 64] = fmaxf(-x, 0.0f);
  } else if (t < 128) {
    int u = t - 64;
    float x = src[64 + u];
    yk[u] = fmaxf(x, 0.0f); yk[u + 64] = fmaxf(-x, 0.0f);
  } else if (t < 192) {
    vv[(size_t)item * DHEAD + (t - 128)] = src[128 + (t - 128)];
  } else if (t == 192) {
    float x = src[192];
    beta[item] = 1.0f / (1.0f + __expf(-x));
  }
  __syncthreads();

  {
    int i = (t >> 7) + 1;     // roll amount 1 or 2
    int tt = t & 127;
    zq[t] = yq[tt] * yq[(tt - i) & 127];
    zk[t] = yk[tt] * yk[(tt - i) & 127];
  }
  __syncthreads();

  red[t] = zq[t]; __syncthreads();
  for (int s = 128; s > 0; s >>= 1) { if (t < s) red[t] += red[t + s]; __syncthreads(); }
  if (t == 0) sums[0] = red[0];
  __syncthreads();
  red[t] = zk[t]; __syncthreads();
  for (int s = 128; s > 0; s >>= 1) { if (t < s) red[t] += red[t + s]; __syncthreads(); }
  if (t == 0) sums[1] = red[0];
  __syncthreads();

  qf[(size_t)item * DK + t] = zq[t] / sums[0];
  kf[(size_t)item * DK + t] = zk[t] / sums[1];
}

// ---------------------------------------------------------------------------
// denominator[l,b,h] = dot(cumsum_l(kf), qf)  — sequential in l per (b,h)
// ---------------------------------------------------------------------------
__global__ __launch_bounds__(256)
void denom_kernel(const float* __restrict__ qf, const float* __restrict__ kf,
                  float* __restrict__ denom) {
  const int bh = blockIdx.x;          // 64 blocks
  const int t = threadIdx.x;          // 256 = DK
  __shared__ float red[256];
  float acc = 0.0f;
  for (int l = 0; l < SLEN; ++l) {
    size_t idx = ((size_t)l * 64 + bh) * DK + t;
    acc += kf[idx];
    red[t] = acc * qf[idx];
    __syncthreads();
    for (int s = 128; s > 0; s >>= 1) { if (t < s) red[t] += red[t + s]; __syncthreads(); }
    if (t == 0) denom[(size_t)l * 64 + bh] = red[0];
    __syncthreads();
  }
}

// ---------------------------------------------------------------------------
// Delta-rule fast-weight recurrence, W[256][64] resident in LDS (64KB) for all
// 2048 steps. One (b,h) per block, thread e owns W column e.
// Writes scaled/normalized output directly as bf16 rows of [ROWS, DMODEL]
// (the A operand of GEMM2).
// ---------------------------------------------------------------------------
__global__ __launch_bounds__(64)
void recurrence_kernel(const float* __restrict__ qf, const float* __restrict__ kf,
                       const float* __restrict__ vv, const float* __restrict__ beta,
                       const float* __restrict__ denom, bf16_t* __restrict__ outbf) {
  extern __shared__ float smem[];
  float* W  = smem;            // 256*64
  float* ks = smem + DK * DHEAD;
  float* qs = ks + DK;

  const int bh = blockIdx.x;   // b*8 + h
  const int b = bh >> 3;
  const int h = bh & 7;
  const int e = threadIdx.x;   // 0..63

  for (int i = e; i < DK * DHEAD; i += 64) W[i] = 0.0f;
  __syncthreads();

  for (int l = 0; l < SLEN; ++l) {
    const size_t base = ((size_t)l * 64 + bh) * DK;
#pragma unroll
    for (int d = e; d < DK; d += 64) { ks[d] = kf[base + d]; qs[d] = qf[base + d]; }
    __syncthreads();

    float v  = vv[((size_t)l * 64 + bh) * DHEAD + e];
    float bt = beta[(size_t)l * 64 + bh];

    float vold = 0.0f;
    for (int d = 0; d < DK; ++d) vold += ks[d] * W[d * DHEAD + e];
    float delta = bt * (v - vold);

    float out = 0.0f;
    for (int d = 0; d < DK; ++d) {
      float w = W[d * DHEAD + e] + ks[d] * delta;
      W[d * DHEAD + e] = w;
      out += qs[d] * w;
    }

    float dn = denom[(size_t)l * 64 + bh];
    out = SCALE_F * out / (dn + EPS_F);
    outbf[(size_t)(l * BSZ + b) * DMODEL + h * DHEAD + e] = (bf16_t)out;
    __syncthreads();
  }
}

// ---------------------------------------------------------------------------
// x = h + attn ; LayerNorm over DMODEL ; one 256-thread block per row
// ---------------------------------------------------------------------------
__global__ __launch_bounds__(256)
void residual_ln_kernel(const float* __restrict__ h, const float* __restrict__ attn,
                        const float* __restrict__ g, const float* __restrict__ bb,
                        float* __restrict__ out) {
  const int row = blockIdx.x;
  const int t = threadIdx.x;
  __shared__ float red[256];
  const size_t o = (size_t)row * DMODEL;
  float x0 = h[o + t]       + attn[o + t];
  float x1 = h[o + 256 + t] + attn[o + 256 + t];

  red[t] = x0 + x1; __syncthreads();
  for (int s = 128; s > 0; s >>= 1) { if (t < s) red[t] += red[t + s]; __syncthreads(); }
  float mu = red[0] * (1.0f / DMODEL);
  __syncthreads();

  float d0 = x0 - mu, d1 = x1 - mu;
  red[t] = d0 * d0 + d1 * d1; __syncthreads();
  for (int s = 128; s > 0; s >>= 1) { if (t < s) red[t] += red[t + s]; __syncthreads(); }
  float inv = rsqrtf(red[0] * (1.0f / DMODEL) + EPS_F);

  out[o + t]       = d0 * inv * g[t]       + bb[t];
  out[o + 256 + t] = d1 * inv * g[t + 256] + bb[t + 256];
}

// ---------------------------------------------------------------------------
extern "C" void kernel_launch(void* const* d_in, const int* in_sizes, int n_in,
                              void* d_out, int out_size, void* d_ws, size_t ws_size,
                              hipStream_t stream) {
  const float* h_f32 = (const float*)d_in[0];   // [2048, 8, 512]
  const float* wqkvb = (const float*)d_in[1];   // [1544, 512]
  const float* wo    = (const float*)d_in[2];   // [512, 512]
  const float* ln_g  = (const float*)d_in[3];   // [512]
  const float* ln_b  = (const float*)d_in[4];   // [512]
  float* out = (float*)d_out;                   // [2048, 8, 512]

  char* w = (char*)d_ws;
  bf16_t* h_bf   = (bf16_t*)w;  w += (size_t)ROWS * DMODEL * 2;
  bf16_t* wq_bf  = (bf16_t*)w;  w += (size_t)EQP * DMODEL * 2;
  bf16_t* wo_bf  = (bf16_t*)w;  w += (size_t)DMODEL * DMODEL * 2;
  float*  qkvb   = (float*)w;   w += (size_t)ROWS * EQ * 4;
  float*  qf     = (float*)w;   w += (size_t)ROWS * NHEAD * DK * 4;
  float*  kf     = (float*)w;   w += (size_t)ROWS * NHEAD * DK * 4;
  float*  vv     = (float*)w;   w += (size_t)ROWS * NHEAD * DHEAD * 4;
  float*  beta   = (float*)w;   w += (size_t)ROWS * NHEAD * 4;
  float*  denom  = (float*)w;   w += (size_t)ROWS * NHEAD * 4;
  bf16_t* out_bf = (bf16_t*)w;  w += (size_t)ROWS * DMODEL * 2;
  float*  attn   = (float*)w;   // ROWS * DMODEL * 4

  // 1) bf16 conversions (W_qkvb padded 1544 -> 1568 rows with zeros)
  {
    size_t n = (size_t)ROWS * DMODEL;
    cvt_pad_bf16<<<dim3((unsigned)((n + 255) / 256)), 256, 0, stream>>>(h_f32, h_bf, ROWS, ROWS, DMODEL);
    n = (size_t)EQP * DMODEL;
    cvt_pad_bf16<<<dim3((unsigned)((n + 255) / 256)), 256, 0, stream>>>(wqkvb, wq_bf, EQ, EQP, DMODEL);
    n = (size_t)DMODEL * DMODEL;
    cvt_pad_bf16<<<dim3((unsigned)((n + 255) / 256)), 256, 0, stream>>>(wo, wo_bf, DMODEL, DMODEL, DMODEL);
  }

  // 2) GEMM1: qkvb[16384,1544] = h[16384,512] x W_qkvb^T   (WMMA bf16)
  gemm_bf16_wmma<<<dim3((EQP / 32 + 7) / 8, ROWS / 16), 256, 0, stream>>>(
      h_bf, wq_bf, qkvb, ROWS, EQ, EQP, DMODEL);

  // 3) DPFP feature map + sigmoid(beta) + v copy
  dpfp_kernel<<<dim3(SLEN * BSZ * NHEAD), 256, 0, stream>>>(qkvb, qf, kf, vv, beta);

  // 4) denominator = dot(cumsum_l(kf), qf)
  denom_kernel<<<dim3(BSZ * NHEAD), 256, 0, stream>>>(qf, kf, denom);

  // 5) LDS-resident delta-rule recurrence -> bf16 out rows
  recurrence_kernel<<<dim3(BSZ * NHEAD), 64, (DK * DHEAD + 2 * DK) * sizeof(float), stream>>>(
      qf, kf, vv, beta, denom, out_bf);

  // 6) GEMM2: attn[16384,512] = out[16384,512] x W_o^T   (WMMA bf16)
  gemm_bf16_wmma<<<dim3((DMODEL / 32) / 8, ROWS / 16), 256, 0, stream>>>(
      out_bf, wo_bf, attn, ROWS, DMODEL, DMODEL, DMODEL);

  // 7) residual + LayerNorm
  residual_ln_kernel<<<dim3(ROWS), 256, 0, stream>>>(h_f32, attn, ln_g, ln_b, out);
}
